// GIN_Graph_33088428049205
// MI455X (gfx1250) — matrix operand order repaired and checked by reading, not compiled, not run on hardware
//
#include <hip/hip_runtime.h>
#include <hip/hip_bf16.h>

typedef __attribute__((ext_vector_type(16))) _Float16 v16h;
typedef __attribute__((ext_vector_type(8)))  float    v8f;

#define GIN_N   100000
#define GIN_E   1600000
#define GIN_G   256
#define GIN_H   32
#define GIN_FIN 128
#define GIN_C   2
#define BN_EPS  1e-5f

// ---------------------------------------------------------------------------
// GEMM1: Z[N x 32] = A[N x K] * W[K x 32], row-major f32 in memory, f16 WMMA
// inputs, f32 accumulation. One wave computes one 16x16 tile of Z.
// A VGPR layout (16-bit A 16x32, contiguous k, same as B): lane<16 -> K = j (0..7) and
// 8+j; lane>=16 -> K = 16+j and 24+j.  a[j] j=0..7 <-> K=hk*16+j,
// a[8+j] <-> K=hk*16+8+j  (hk = lane>>4).
// B VGPR layout (32x16): lanes 0-15 hold K=0..15, lanes 16-31 hold K=16..31,
// element b[j] <-> K = hk*16 + j, column = lane&15.
// D layout: VGPR r, lane l -> row = r + 8*(l>>4), col = l&15.
// ---------------------------------------------------------------------------
__global__ void gin_gemm1(const float* __restrict__ A, const float* __restrict__ W,
                          float* __restrict__ Z, int n_rows, int K) {
  int wave = (int)((blockIdx.x * blockDim.x + threadIdx.x) >> 5);
  int lane = threadIdx.x & 31;
  int total = (n_rows >> 4) * 2;           // (N/16) row tiles x 2 col tiles
  if (wave >= total) return;               // wave-uniform branch: EXEC stays all-1
  int m0 = (wave >> 1) << 4;
  int n0 = (wave & 1) << 4;
  int lr = lane & 15;
  int hk = lane >> 4;

  const float* arow = A + (size_t)(m0 + lr) * K;
  const float* wcol = W + (n0 + lr);

  v8f acc = {};
  for (int k0 = 0; k0 < K; k0 += 32) {
    if (k0 + 32 < K) __builtin_prefetch(arow + k0 + 32, 0, 0);  // global_prefetch_b8
    const float* ap = arow + k0;
    const float* bp = wcol + (size_t)(k0 + hk * 16) * GIN_H;
    v16h a, b;
#pragma unroll
    for (int j = 0; j < 8; ++j) {
      a[j]     = (_Float16)ap[hk * 16 + j];
      a[8 + j] = (_Float16)ap[hk * 16 + 8 + j];
    }
#pragma unroll
    for (int j = 0; j < 16; ++j) {
      b[j] = (_Float16)bp[(size_t)j * GIN_H];
    }
    acc = __builtin_amdgcn_wmma_f32_16x16x32_f16(false, a, false, b,
                                                 (short)0, acc, false, false);
  }
  float* zp = Z + (size_t)(m0 + 8 * hk) * GIN_H + n0 + lr;
#pragma unroll
  for (int r = 0; r < 8; ++r) zp[(size_t)r * GIN_H] = acc[r];
}

// ---------------------------------------------------------------------------
// GEMM2 (K=32) with fused epilogue: V = relu(U @ W + bias); also accumulates
// per-feature sum / sum-of-squares into stats[0..31] / stats[32..63] for BN
// (each lane owns a single output column -> 2 atomics per lane).
// ---------------------------------------------------------------------------
__global__ void gin_gemm2(const float* __restrict__ U, const float* __restrict__ W,
                          const float* __restrict__ bias, float* __restrict__ V,
                          float* __restrict__ stats, int n_rows) {
  int wave = (int)((blockIdx.x * blockDim.x + threadIdx.x) >> 5);
  int lane = threadIdx.x & 31;
  int total = (n_rows >> 4) * 2;
  if (wave >= total) return;
  int m0 = (wave >> 1) << 4;
  int n0 = (wave & 1) << 4;
  int lr = lane & 15;
  int hk = lane >> 4;

  const float* ap = U + (size_t)(m0 + lr) * GIN_H;
  const float* bp = W + (size_t)(hk * 16) * GIN_H + n0 + lr;

  v16h a, b;
#pragma unroll
  for (int j = 0; j < 8; ++j) {
    a[j]     = (_Float16)ap[hk * 16 + j];
    a[8 + j] = (_Float16)ap[hk * 16 + 8 + j];
  }
#pragma unroll
  for (int j = 0; j < 16; ++j) b[j] = (_Float16)bp[(size_t)j * GIN_H];

  v8f acc = {};
  acc = __builtin_amdgcn_wmma_f32_16x16x32_f16(false, a, false, b,
                                               (short)0, acc, false, false);

  int col = n0 + lr;
  float bv = bias[col];
  float s = 0.f, sq = 0.f;
  float* vp = V + (size_t)(m0 + 8 * hk) * GIN_H + col;
#pragma unroll
  for (int r = 0; r < 8; ++r) {
    float v = fmaxf(acc[r] + bv, 0.f);
    vp[(size_t)r * GIN_H] = v;
    s += v;
    sq += v * v;
  }
  atomicAdd(&stats[col], s);
  atomicAdd(&stats[GIN_H + col], sq);
}

// agg[dst] += z[src] over all (edge, feature) pairs; 32 lanes cover one edge's
// feature row -> coalesced 128B gather, float atomic scatter.
__global__ void gin_scatter(const int* __restrict__ src, const int* __restrict__ dst,
                            const float* __restrict__ z, float* __restrict__ agg,
                            int total) {
  int i = (int)(blockIdx.x * blockDim.x + threadIdx.x);
  if (i >= total) return;
  int e = i >> 5, f = i & 31;
  atomicAdd(&agg[(size_t)dst[e] * GIN_H + f], z[(size_t)src[e] * GIN_H + f]);
}

__global__ void gin_copy4(const float4* __restrict__ s, float4* __restrict__ d, int n) {
  int i = (int)(blockIdx.x * blockDim.x + threadIdx.x);
  if (i < n) d[i] = s[i];
}

__global__ void gin_zero(float* __restrict__ p, int n) {
  int i = (int)(blockIdx.x * blockDim.x + threadIdx.x);
  if (i < n) p[i] = 0.f;
}

// u = relu(agg + b1)  (in place)
__global__ void gin_bias_relu(float* __restrict__ a, const float* __restrict__ b, int total) {
  int i = (int)(blockIdx.x * blockDim.x + threadIdx.x);
  if (i < total) a[i] = fmaxf(a[i] + b[i & 31], 0.f);
}

// stats[f] = scale*rsqrt(var+eps); stats[32+f] = bias - mean*that (in place).
__global__ void gin_bn_final(float* __restrict__ stats, const float* __restrict__ g,
                             const float* __restrict__ bb, float inv_n) {
  int f = threadIdx.x;
  if (f < GIN_H) {
    float mean = stats[f] * inv_n;
    float var = stats[GIN_H + f] * inv_n - mean * mean;
    float aa = g[f] * rsqrtf(var + BN_EPS);
    stats[f] = aa;
    stats[GIN_H + f] = bb[f] - mean * aa;
  }
}

// h = h*a + b (in place)
__global__ void gin_bn_apply(float* __restrict__ h, const float* __restrict__ ab, int total) {
  int i = (int)(blockIdx.x * blockDim.x + threadIdx.x);
  if (i < total) {
    int f = i & 31;
    h[i] = h[i] * ab[f] + ab[GIN_H + f];
  }
}

// pooled[batch[n]] += h[n]
__global__ void gin_pool(const float* __restrict__ h, const int* __restrict__ batch,
                         float* __restrict__ pooled, int total) {
  int i = (int)(blockIdx.x * blockDim.x + threadIdx.x);
  if (i >= total) return;
  int n = i >> 5, f = i & 31;
  atomicAdd(&pooled[(size_t)batch[n] * GIN_H + f], h[i]);
}

// Head: per-graph fc1(relu) -> fc2 -> log_softmax (C=2). One thread per graph.
__global__ void gin_head(const float* __restrict__ pooled,
                         const float* __restrict__ w1, const float* __restrict__ b1,
                         const float* __restrict__ w2, const float* __restrict__ b2,
                         float* __restrict__ out) {
  int g = threadIdx.x;
  if (g >= GIN_G) return;
  float p[GIN_H], hr[GIN_H];
#pragma unroll
  for (int k = 0; k < GIN_H; ++k) p[k] = pooled[(size_t)g * GIN_H + k];
#pragma unroll
  for (int o = 0; o < GIN_H; ++o) {
    float acc = b1[o];
    for (int k = 0; k < GIN_H; ++k) acc += p[k] * w1[k * GIN_H + o];
    hr[o] = fmaxf(acc, 0.f);
  }
  float l0 = b2[0], l1 = b2[1];
  for (int k = 0; k < GIN_H; ++k) {
    l0 += hr[k] * w2[k * GIN_C + 0];
    l1 += hr[k] * w2[k * GIN_C + 1];
  }
  float m = fmaxf(l0, l1);
  float lse = m + logf(__expf(l0 - m) + __expf(l1 - m));
  out[(size_t)g * GIN_C + 0] = l0 - lse;
  out[(size_t)g * GIN_C + 1] = l1 - lse;
}

extern "C" void kernel_launch(void* const* d_in, const int* in_sizes, int n_in,
                              void* d_out, int out_size, void* d_ws, size_t ws_size,
                              hipStream_t stream) {
  (void)in_sizes; (void)n_in; (void)out_size; (void)ws_size;

  const float* x     = (const float*)d_in[0];
  const int*   ei    = (const int*)d_in[1];     // (2,E): [0..E) = src, [E..2E) = dst
  const int*   batch = (const int*)d_in[2];
  const float* w1_0  = (const float*)d_in[3];
  const float* b1_0  = (const float*)d_in[4];
  const float* w2_0  = (const float*)d_in[5];
  const float* b2_0  = (const float*)d_in[6];
  const float* ws1   = (const float*)d_in[7];   // (3,32,32)
  const float* bs1   = (const float*)d_in[8];   // (3,32)
  const float* ws2   = (const float*)d_in[9];
  const float* bs2   = (const float*)d_in[10];
  const float* bn_g  = (const float*)d_in[11];  // (4,32)
  const float* bn_b  = (const float*)d_in[12];
  const float* fc1w  = (const float*)d_in[13];
  const float* fc1b  = (const float*)d_in[14];
  const float* fc2w  = (const float*)d_in[15];
  const float* fc2b  = (const float*)d_in[16];
  float* out = (float*)d_out;

  const size_t NH = (size_t)GIN_N * GIN_H;      // 3.2M floats
  float* buf[3];
  buf[0] = (float*)d_ws;
  buf[1] = buf[0] + NH;
  buf[2] = buf[1] + NH;
  float* stats  = buf[2] + NH;                  // 64 floats
  float* pooled = stats + 2 * GIN_H;            // G*H floats

  const int THR = 256;
  const int elemsNH = GIN_N * GIN_H;            // 3,200,000
  const int blkNH   = (elemsNH + THR - 1) / THR;
  const int elemsE  = GIN_E * GIN_H;            // 51,200,000
  const int blkE    = (elemsE + THR - 1) / THR;
  const int nCopy4  = elemsNH / 4;
  const int blkC4   = (nCopy4 + THR - 1) / THR;
  const int tiles   = (GIN_N / 16) * 2;         // 12500 waves
  const int blkGEMM = (tiles * 32 + THR - 1) / THR;

  const int* src = ei;
  const int* dst = ei + GIN_E;

  for (int layer = 0; layer < 4; ++layer) {
    const float* hin = (layer == 0) ? x : buf[(layer - 1) % 3];
    int K            = (layer == 0) ? GIN_FIN : GIN_H;
    const float* w1  = (layer == 0) ? w1_0 : ws1 + (size_t)(layer - 1) * GIN_H * GIN_H;
    const float* b1  = (layer == 0) ? b1_0 : bs1 + (size_t)(layer - 1) * GIN_H;
    const float* w2  = (layer == 0) ? w2_0 : ws2 + (size_t)(layer - 1) * GIN_H * GIN_H;
    const float* b2  = (layer == 0) ? b2_0 : bs2 + (size_t)(layer - 1) * GIN_H;
    float* z   = buf[layer % 3];
    float* agg = buf[(layer + 1) % 3];

    // z = hin @ w1   (project BEFORE aggregating: GIN MLP first op is linear)
    gin_gemm1<<<blkGEMM, THR, 0, stream>>>(hin, w1, z, GIN_N, K);
    // agg = z  (self term)
    gin_copy4<<<blkC4, THR, 0, stream>>>((const float4*)z, (float4*)agg, nCopy4);
    // agg[dst] += z[src]
    gin_scatter<<<blkE, THR, 0, stream>>>(src, dst, z, agg, elemsE);
    // u = relu(agg + b1)
    gin_bias_relu<<<blkNH, THR, 0, stream>>>(agg, b1, elemsNH);
    // BN stats accumulators
    gin_zero<<<1, 64, 0, stream>>>(stats, 2 * GIN_H);
    // v = relu(u @ w2 + b2), fused BN sum/sumsq
    gin_gemm2<<<blkGEMM, THR, 0, stream>>>(agg, w2, b2, z, stats, GIN_N);
    // fold mean/var + affine into (a,b)
    gin_bn_final<<<1, GIN_H, 0, stream>>>(stats, bn_g + (size_t)layer * GIN_H,
                                          bn_b + (size_t)layer * GIN_H,
                                          1.0f / (float)GIN_N);
    // h = v*a + b  (in place -> layer output lives in z's buffer)
    gin_bn_apply<<<blkNH, THR, 0, stream>>>(z, stats, elemsNH);
  }

  float* h_final = buf[3 % 3];  // layer 3 output -> buf[0]
  gin_zero<<<(GIN_G * GIN_H + THR - 1) / THR, THR, 0, stream>>>(pooled, GIN_G * GIN_H);
  gin_pool<<<blkNH, THR, 0, stream>>>(h_final, batch, pooled, elemsNH);
  gin_head<<<1, GIN_G, 0, stream>>>(pooled, fc1w, fc1b, fc2w, fc2b, out);
}
